// AttnLayer_26671746908719
// MI455X (gfx1250) — matrix operand order repaired and checked by reading, not compiled, not run on hardware
//
#include <hip/hip_runtime.h>
#include <hip/hip_bf16.h>
#include <math.h>

#define NB    32      // batch
#define TN    8192    // T*N history slots per batch
#define HID   128
#define ATTNA 128

typedef __attribute__((ext_vector_type(16))) __bf16 v16bf;
typedef __attribute__((ext_vector_type(8)))  float  v8f;

// round-to-nearest-even pack of two f32 into one dword of two bf16
__device__ __forceinline__ unsigned int pack2_bf16(float lo, float hi) {
    unsigned int a = __float_as_uint(lo);
    unsigned int b = __float_as_uint(hi);
    a += 0x7FFFu + ((a >> 16) & 1u);
    b += 0x7FFFu + ((b >> 16) & 1u);
    return (a >> 16) | (b & 0xFFFF0000u);
}

// branchless tanh: one v_exp_f32 + one raw v_rcp_f32, no division chain,
// no exec-mask divergence. |err| ~1 ulp of rcp -> far below bf16 noise.
__device__ __forceinline__ float fast_tanh(float x) {
    float ax = fabsf(x);
    float e  = __expf(-2.0f * ax);                       // (0,1]
    float r  = (1.0f - e) * __builtin_amdgcn_rcpf(1.0f + e);
    return copysignf(r, x);
}

// ---------------------------------------------------------------------------
// Kernel 0: add_term[b][a] = cur_h[b] . Wx_w[a] + Wx_b[a] + Wh_b[a]
// ---------------------------------------------------------------------------
__global__ __launch_bounds__(128)
void prep_kernel(const float* __restrict__ cur, const float* __restrict__ Wx_w,
                 const float* __restrict__ Wx_b, const float* __restrict__ Wh_b,
                 float* __restrict__ add_term) {
    const int b = blockIdx.x, a = threadIdx.x;
    __shared__ float lc[HID];
    lc[a] = cur[b * HID + a];
    __syncthreads();
    float acc = Wx_b[a] + Wh_b[a];
    const float* wr = Wx_w + a * HID;
#pragma unroll 8
    for (int k = 0; k < HID; ++k) acc += lc[k] * wr[k];
    add_term[b * ATTNA + a] = acc;
}

// ---------------------------------------------------------------------------
// Kernel 1: scores via bf16 WMMA, double-buffered LDS staging.
// Grid (32 chunks, 32 batches), 256 threads (8 waves). Wave w owns A-columns
// [16w,16w+16). Each block processes 256 rows as 16 tiles of 16 rows.
// ---------------------------------------------------------------------------
__global__ __launch_bounds__(256)
void score_kernel(const float* __restrict__ hist, const float* __restrict__ Wh_w,
                  const float* __restrict__ add_term, const float* __restrict__ v_w,
                  float* __restrict__ score) {
    const int tid   = threadIdx.x;
    const int lane  = tid & 31;
    const int w     = tid >> 5;           // wave id == A column tile
    const int b     = blockIdx.y;
    const int chunk = blockIdx.x;

    __shared__ __align__(16) unsigned int ldsA[2][16 * 64];  // 2 x (16 rows x 128 bf16)
    __shared__ float lds_partial[8 * 16];

    const int   acol   = w * 16 + (lane & 15);
    const float addt   = add_term[b * ATTNA + acol];
    const float vw     = v_w[acol];
    const int   row    = lane & 15;
    const int   hiHalf = (lane >> 4) & 1;

    // B fragments: B[k][a] = Wh_w[a][k], bf16, contiguous k order (k = 16h + e), same as A.
    // lanes 0-15 carry K = 32s+0..15 (2/VGPR), lanes 16-31 carry K = 32s+16..31.
    unsigned int bfr[4][8];
#pragma unroll
    for (int s = 0; s < 4; ++s) {
        const int kbase = 32 * s + (hiHalf ? 16 : 0);
        const float4* wq = (const float4*)(Wh_w + acol * HID + kbase);
        float4 q0 = wq[0], q1 = wq[1], q2 = wq[2], q3 = wq[3];
        bfr[s][0] = pack2_bf16(q0.x, q0.y); bfr[s][1] = pack2_bf16(q0.z, q0.w);
        bfr[s][2] = pack2_bf16(q1.x, q1.y); bfr[s][3] = pack2_bf16(q1.z, q1.w);
        bfr[s][4] = pack2_bf16(q2.x, q2.y); bfr[s][5] = pack2_bf16(q2.z, q2.w);
        bfr[s][6] = pack2_bf16(q3.x, q3.y); bfr[s][7] = pack2_bf16(q3.z, q3.w);
    }

    const size_t rowBase = (size_t)b * TN + (size_t)chunk * 256;

    // preload tile 0 (16x128 f32, coalesced, 8 floats/thread)
    float4 x0, x1;
    {
        const float4* src = (const float4*)(hist + rowBase * HID) + tid * 2;
        x0 = src[0]; x1 = src[1];
    }

    for (int t = 0; t < 16; ++t) {
        unsigned int* bufA = ldsA[t & 1];
        // convert staged tile to bf16 and publish in LDS
        unsigned int p0 = pack2_bf16(x0.x, x0.y), p1 = pack2_bf16(x0.z, x0.w);
        unsigned int p2 = pack2_bf16(x1.x, x1.y), p3 = pack2_bf16(x1.z, x1.w);
        *(uint4*)&bufA[tid * 4] = make_uint4(p0, p1, p2, p3);
        __syncthreads();

        // kick off next tile's global loads; they overlap the WMMA work below
        if (t < 15) {
            const float4* src = (const float4*)(hist + (rowBase + (t + 1) * 16) * HID) + tid * 2;
            x0 = src[0]; x1 = src[1];
        }

        // fetch all A fragments, then 4 chained WMMAs (K = 128)
        uint4 lo[4], hi[4];
#pragma unroll
        for (int s = 0; s < 4; ++s) {
            const unsigned int off = row * 64 + s * 16 + (hiHalf ? 8u : 0u);
            lo[s] = *(const uint4*)&bufA[off];
            hi[s] = *(const uint4*)&bufA[off + 4];
        }

        v8f c = {0.f, 0.f, 0.f, 0.f, 0.f, 0.f, 0.f, 0.f};
#pragma unroll
        for (int s = 0; s < 4; ++s) {
            union { unsigned int u[8]; v16bf v; } afr, bfu;
            afr.u[0] = lo[s].x; afr.u[1] = lo[s].y; afr.u[2] = lo[s].z; afr.u[3] = lo[s].w;
            afr.u[4] = hi[s].x; afr.u[5] = hi[s].y; afr.u[6] = hi[s].z; afr.u[7] = hi[s].w;
#pragma unroll
            for (int j = 0; j < 8; ++j) bfu.u[j] = bfr[s][j];
            c = __builtin_amdgcn_wmma_f32_16x16x32_bf16(false, afr.v, false, bfu.v,
                                                        (short)0, c, false, false);
        }

        // score contribution: sum_a tanh(proj+add)*v over this wave's 16 columns
        float tv[8];
#pragma unroll
        for (int j = 0; j < 8; ++j) {
            float x = fast_tanh(c[j] + addt) * vw;
#pragma unroll
            for (int m = 1; m < 16; m <<= 1) x += __shfl_xor(x, m, 32);
            tv[j] = x;   // row (j + 8*hiHalf) partial sum, replicated in half
        }
        if ((lane & 15) == 0) {
            const int rb = hiHalf * 8;
#pragma unroll
            for (int j = 0; j < 8; ++j) lds_partial[w * 16 + rb + j] = tv[j];
        }
        __syncthreads();
        if (tid < 16) {
            float sacc = 0.f;
#pragma unroll
            for (int w2 = 0; w2 < 8; ++w2) sacc += lds_partial[w2 * 16 + tid];
            score[rowBase + t * 16 + tid] = sacc;
        }
        // next writes to lds_partial / this LDS buffer are ordered behind the
        // next iteration's first barrier -> no third barrier needed.
    }
}

// ---------------------------------------------------------------------------
// Kernel 2: softmax over 8192 slots per batch (one block/batch, in place)
// ---------------------------------------------------------------------------
__global__ __launch_bounds__(256)
void softmax_kernel(float* __restrict__ attn) {
    const int b = blockIdx.x, tid = threadIdx.x;
    __shared__ float red[256];
    float* p = attn + (size_t)b * TN;

    float m = -3.402823466e38f;
    for (int i = tid; i < TN; i += 256) m = fmaxf(m, p[i]);
    red[tid] = m; __syncthreads();
    for (int s = 128; s > 0; s >>= 1) {
        if (tid < s) red[tid] = fmaxf(red[tid], red[tid + s]);
        __syncthreads();
    }
    m = red[0]; __syncthreads();

    float ssum = 0.f;
    for (int i = tid; i < TN; i += 256) {
        float e = __expf(p[i] - m);
        p[i] = e;
        ssum += e;
    }
    red[tid] = ssum; __syncthreads();
    for (int s = 128; s > 0; s >>= 1) {
        if (tid < s) red[tid] += red[tid + s];
        __syncthreads();
    }
    const float inv = 1.0f / red[0];
    for (int i = tid; i < TN; i += 256) p[i] *= inv;
}

// ---------------------------------------------------------------------------
// Kernel 3: partial[b][chunk][h] = sum over 256 rows of attn[s]*hist[s][h]
// (second history pass; mostly L2-resident after score_kernel)
// ---------------------------------------------------------------------------
__global__ __launch_bounds__(128)
void wsum_kernel(const float* __restrict__ hist, const float* __restrict__ attn,
                 float* __restrict__ partial) {
    const int b = blockIdx.y, ck = blockIdx.x, h = threadIdx.x;
    __shared__ float la[256];
    la[h]       = attn[(size_t)b * TN + ck * 256 + h];
    la[h + 128] = attn[(size_t)b * TN + ck * 256 + 128 + h];
    __syncthreads();
    const float* hp = hist + ((size_t)b * TN + ck * 256) * HID + h;
    float acc = 0.f;
#pragma unroll
    for (int r0 = 0; r0 < 256; r0 += 64) {
        if (r0 + 64 < 256)  // in-bounds stripe prefetch -> global_prefetch_b8
            __builtin_prefetch(hp + (size_t)(r0 + 64) * HID, 0, 0);
#pragma unroll
        for (int r = r0; r < r0 + 64; ++r)
            acc += la[r] * hp[(size_t)r * HID];
    }
    partial[((size_t)b * 32 + ck) * HID + h] = acc;
}

// ---------------------------------------------------------------------------
// Kernel 4: out[b][h] = cur_h[b][h] + sum_c partial[b][c][h]  (fixed order)
// ---------------------------------------------------------------------------
__global__ __launch_bounds__(128)
void finish_kernel(const float* __restrict__ cur, const float* __restrict__ partial,
                   float* __restrict__ out) {
    const int b = blockIdx.x, h = threadIdx.x;
    float acc = cur[b * HID + h];
#pragma unroll
    for (int c = 0; c < 32; ++c) acc += partial[((size_t)b * 32 + c) * HID + h];
    out[b * HID + h] = acc;
}

// ---------------------------------------------------------------------------
extern "C" void kernel_launch(void* const* d_in, const int* in_sizes, int n_in,
                              void* d_out, int out_size, void* d_ws, size_t ws_size,
                              hipStream_t stream) {
    const float* cur  = (const float*)d_in[0];   // [32][128]
    const float* hist = (const float*)d_in[1];   // [32][8192][128]
    const float* Wx_w = (const float*)d_in[2];   // [128][128]
    const float* Wx_b = (const float*)d_in[3];   // [128]
    const float* Wh_w = (const float*)d_in[4];   // [128][128]
    const float* Wh_b = (const float*)d_in[5];   // [128]
    const float* v_w  = (const float*)d_in[6];   // [128]
    float* out = (float*)d_out;

    float* ws       = (float*)d_ws;
    float* add_term = ws;                         // 32*128      = 4096
    float* attn     = ws + 4096;                  // 32*8192     = 262144
    float* partial  = ws + 4096 + 262144;         // 32*32*128   = 131072

    prep_kernel   <<<dim3(NB),      dim3(128), 0, stream>>>(cur, Wx_w, Wx_b, Wh_b, add_term);
    score_kernel  <<<dim3(32, NB),  dim3(256), 0, stream>>>(hist, Wh_w, add_term, v_w, attn);
    softmax_kernel<<<dim3(NB),      dim3(256), 0, stream>>>(attn);
    wsum_kernel   <<<dim3(32, NB),  dim3(128), 0, stream>>>(hist, attn, partial);
    finish_kernel <<<dim3(NB),      dim3(128), 0, stream>>>(cur, partial, out);
}
